// SKCLite3D_10359461118249
// MI455X (gfx1250) — matrix operand order repaired and checked by reading, not compiled, not run on hardware
//
#include <hip/hip_runtime.h>
#include <hip/hip_bf16.h>
#include <math.h>

// ---------------------------------------------------------------------------
// SKC-Lite 3D dual cross-attention for gfx1250 (MI455X).
// B=2, C=256, N=4096, R=8, G=4.
//
// Pipeline (softmax is separable: attn[n,m] = exp(s[n,m]) / Z[n]):
//   1) proj8:   q,k = W x + b            (fp32 AoS [n][8])
//   2) ekernel: S tiles via 2x v_wmma_f32_16x16x4_f32 (K=8 fp32), E=exp(S)
//               stored transposed bf16 E[m][n]; Z[n]=sum_m E via xor-shuffles
//   3) gconvZ:  V[c,n] = (grouped conv + bias) * (1/Z[n])  -> bf16
//   4) attn:    out = f + beta * (V x E^T) -- pure bf16 WMMA GEMM.
//               Per wave: 32x64 output tile (2 c-tiles x 4 m-tiles) ->
//               12 b128 loads per 8 WMMAs (1.5 loads/WMMA, ~21 FLOP/byte).
// ---------------------------------------------------------------------------

#define Ncfg 4096
#define Ccfg 256
#define Rcfg 8

typedef __attribute__((ext_vector_type(16))) __bf16 v16bf;
typedef __attribute__((ext_vector_type(8)))  __bf16 bf16x8;
typedef __attribute__((ext_vector_type(8)))  float  v8f;
typedef __attribute__((ext_vector_type(2)))  float  v2f;

union ABFrag { v16bf v; bf16x8 h[2]; };

// ---------------------------------------------------------------------------
// Kernel 1: rank-8 1x1x1 projection  out[n][r] = bias[r] + sum_c W[r][c]*X[c][n]
// ---------------------------------------------------------------------------
__global__ __launch_bounds__(256) void proj8_kernel(
    const float* __restrict__ X, const float* __restrict__ W,
    const float* __restrict__ bias, float* __restrict__ out) {
  __shared__ float wT[Ccfg * Rcfg];  // [c][r]
  __shared__ float bsh[Rcfg];
  const int tid = threadIdx.x;
  for (int i = tid; i < Rcfg * Ccfg; i += 256) {
    int r = i / Ccfg, c = i % Ccfg;
    wT[c * Rcfg + r] = W[i];
  }
  if (tid < Rcfg) bsh[tid] = bias[tid];
  __syncthreads();

  const int n = blockIdx.x * 256 + tid;
  float acc[Rcfg];
#pragma unroll
  for (int r = 0; r < Rcfg; ++r) acc[r] = bsh[r];
  for (int c = 0; c < Ccfg; ++c) {
    const float x = X[(size_t)c * Ncfg + n];
#pragma unroll
    for (int r = 0; r < Rcfg; ++r) acc[r] += wT[c * Rcfg + r] * x;
  }
  float4* o = (float4*)(out + (size_t)n * Rcfg);
  o[0] = make_float4(acc[0], acc[1], acc[2], acc[3]);
  o[1] = make_float4(acc[4], acc[5], acc[6], acc[7]);
}

// ---------------------------------------------------------------------------
// Kernel 2: E[m][n] = exp(q[n].k[m]) (bf16, transposed) + invZ[n] = 1/sum_m E.
// Logit GEMM on the matrix pipe: 2x chained V_WMMA_F32_16X16X4_F32 (fp32, K=8).
// D layout: element r of a lane = (n = n0 + r + rh, m = mtile + lane&15) ->
// each lane stores 8 consecutive n's of one m-row: one 16B store of bf16x8.
// grid = (N/16 n-tiles, 4 instances), block = 256 (8 waves, each 32 m-tiles).
// ---------------------------------------------------------------------------
__global__ __launch_bounds__(256) void ekernel(
    const float* __restrict__ qbuf, const float* __restrict__ kbuf,
    __bf16* __restrict__ Ebuf, float* __restrict__ invZbuf) {
  __shared__ float zpart[8][16];
  const int tid = threadIdx.x;
  const int lane = tid & 31, wave = tid >> 5;
  const int inst = blockIdx.y;
  const int n0 = blockIdx.x * 16;
  const int row = lane & 15;
  const int rh = (lane < 16) ? 0 : 8;   // D-layout row offset for this half
  const int ksel = (lane < 16) ? 0 : 2; // 32-bit A/B: K {0,1}|{2,3} per half

  const float* qb = qbuf + (size_t)inst * Ncfg * Rcfg;
  const float* kb = kbuf + (size_t)inst * Ncfg * Rcfg;
  __bf16* E = Ebuf + (size_t)inst * Ncfg * Ncfg;
  float* invZ = invZbuf + (size_t)inst * Ncfg;

  const int n = n0 + row;
  const v2f qa = *(const v2f*)(qb + (size_t)n * Rcfg + ksel);      // K 0..3
  const v2f qc = *(const v2f*)(qb + (size_t)n * Rcfg + 4 + ksel);  // K 4..7

  float zacc[8];
#pragma unroll
  for (int r = 0; r < 8; ++r) zacc[r] = 0.0f;

  for (int mt = wave; mt < Ncfg / 16; mt += 8) {
    const int m = mt * 16 + row;
    const v2f ka = *(const v2f*)(kb + (size_t)m * Rcfg + ksel);
    const v2f kc = *(const v2f*)(kb + (size_t)m * Rcfg + 4 + ksel);
    v8f s = {};
    s = __builtin_amdgcn_wmma_f32_16x16x4_f32(false, qa, false, ka, (short)0,
                                              s, false, false);
    s = __builtin_amdgcn_wmma_f32_16x16x4_f32(false, qc, false, kc, (short)0,
                                              s, false, false);
    bf16x8 e;
#pragma unroll
    for (int r = 0; r < 8; ++r) {
      const float ev = __expf(s[r]);
      zacc[r] += ev;
      e[r] = (__bf16)ev;
    }
    *(bf16x8*)(E + (size_t)m * Ncfg + n0 + rh) = e;  // 16B store, aligned
  }

  // Z[n]: reduce over m. xor-shuffles (1,2,4,8) stay within each 16-lane half.
#pragma unroll
  for (int off = 1; off < 16; off <<= 1) {
#pragma unroll
    for (int r = 0; r < 8; ++r) zacc[r] += __shfl_xor(zacc[r], off, 32);
  }
  if (lane == 0) {
#pragma unroll
    for (int r = 0; r < 8; ++r) zpart[wave][r] = zacc[r];
  }
  if (lane == 16) {
#pragma unroll
    for (int r = 0; r < 8; ++r) zpart[wave][8 + r] = zacc[r];
  }
  __syncthreads();
  if (tid < 16) {
    float z = 0.0f;
#pragma unroll
    for (int w = 0; w < 8; ++w) z += zpart[w][tid];
    invZ[n0 + tid] = 1.0f / z;
  }
}

// ---------------------------------------------------------------------------
// Kernel 3: grouped 1x1x1 conv with 1/Z[n] folded in -> bf16 V buffer.
// ---------------------------------------------------------------------------
__global__ __launch_bounds__(256) void gconvz_kernel(
    const float* __restrict__ X, const float* __restrict__ W,
    const float* __restrict__ bias, const float* __restrict__ invZ,
    __bf16* __restrict__ out) {
  __shared__ float wsh[64];
  const int c = blockIdx.y;
  const int g = c >> 6, co = c & 63;
  const int tid = threadIdx.x;
  if (tid < 64) wsh[tid] = W[((size_t)g * 64 + co) * 64 + tid];
  __syncthreads();

  const int n = blockIdx.x * 256 + tid;
  float acc = bias[c];
  const float* xg = X + (size_t)(g * 64) * Ncfg + n;
#pragma unroll 8
  for (int i = 0; i < 64; ++i) acc += wsh[i] * xg[(size_t)i * Ncfg];
  out[(size_t)c * Ncfg + n] = (__bf16)(acc * invZ[n]);
}

// ---------------------------------------------------------------------------
// Kernel 4: out = f_in + beta * (V x E^T). Pure bf16 WMMA GEMM, no LDS.
// grid = (N/64 m-tiles, 4 instances), block = 256 (8 waves x 32 c-rows).
// Each wave: 32x64 output tile = 2 c-tiles x 4 m-tiles (8 v8f accumulators).
// Per 32-K chunk per lane: 12x global_load_b128 feeding 8 WMMAs.
// ---------------------------------------------------------------------------
__global__ __launch_bounds__(256) void attn_gemm_kernel(
    const __bf16* __restrict__ vbuf, const __bf16* __restrict__ Ebuf,
    const float* __restrict__ fL, const float* __restrict__ fU,
    float* __restrict__ out, const float* __restrict__ beta_ptr) {
  const int tid = threadIdx.x;
  const int lane = tid & 31, wave = tid >> 5;
  const int inst = blockIdx.y;
  const int dir = inst >> 1;  // 0: LU (updates fL), 1: UL (updates fU)
  const int b = inst & 1;
  const int m0 = blockIdx.x * 64;

  const __bf16* vb = vbuf + (size_t)inst * Ccfg * Ncfg;
  const __bf16* E = Ebuf + (size_t)inst * Ncfg * Ncfg;

  const int aoff = (lane < 16) ? 0 : 8;   // 16-bit A: K {0-7,16-23}|{8-15,24-31}
  const int boff = (lane < 16) ? 0 : 8;   // 16-bit B: K {0-7,16-23}|{8-15,24-31}
  const __bf16* bp = E + (size_t)(m0 + (lane & 15)) * Ncfg + boff;
  const __bf16* ap = vb + (size_t)(wave * 32 + (lane & 15)) * Ncfg + aoff;

  v8f acc[2][4] = {};

  for (int n0 = 0; n0 < Ncfg; n0 += 32) {
    ABFrag bfr[4];
#pragma unroll
    for (int mt = 0; mt < 4; ++mt) {
      const __bf16* bpt = bp + (size_t)(mt * 16) * Ncfg + n0;
      bfr[mt].h[0] = *(const bf16x8*)(bpt);       // K 0..7   (| 8..15)
      bfr[mt].h[1] = *(const bf16x8*)(bpt + 16);  // K 16..23 (| 24..31)
    }
#pragma unroll
    for (int ct = 0; ct < 2; ++ct) {
      const __bf16* apt = ap + (size_t)(ct * 16) * Ncfg + n0;
      ABFrag af;
      af.h[0] = *(const bf16x8*)(apt);
      af.h[1] = *(const bf16x8*)(apt + 16);
#pragma unroll
      for (int mt = 0; mt < 4; ++mt)
        acc[ct][mt] = __builtin_amdgcn_wmma_f32_16x16x32_bf16(
            false, af.v, false, bfr[mt].v, (short)0, acc[ct][mt], false, false);
    }
  }

  // Epilogue: f32 C/D layout -> out = f_in + beta*acc.
  const float beta = *beta_ptr;
  const size_t CN = (size_t)Ccfg * Ncfg;
  const float* fin = (dir ? fU : fL) + (size_t)b * CN;
  float* op = out + (size_t)dir * 2 * CN + (size_t)b * CN;
  const int rowadd = (lane < 16) ? 0 : 8;
#pragma unroll
  for (int ct = 0; ct < 2; ++ct) {
#pragma unroll
    for (int mt = 0; mt < 4; ++mt) {
      const int mcol = m0 + mt * 16 + (lane & 15);
#pragma unroll
      for (int r = 0; r < 8; ++r) {
        const int c = wave * 32 + ct * 16 + r + rowadd;
        const size_t i = (size_t)c * Ncfg + mcol;
        op[i] = fin[i] + beta * acc[ct][mt][r];
      }
    }
  }
}

// ---------------------------------------------------------------------------
// Host launch
// ---------------------------------------------------------------------------
extern "C" void kernel_launch(void* const* d_in, const int* in_sizes, int n_in,
                              void* d_out, int out_size, void* d_ws, size_t ws_size,
                              hipStream_t stream) {
  (void)in_sizes; (void)n_in; (void)out_size; (void)ws_size;
  const float* fL   = (const float*)d_in[0];
  const float* fU   = (const float*)d_in[1];
  const float* qL_w = (const float*)d_in[2];
  const float* qL_b = (const float*)d_in[3];
  const float* kU_w = (const float*)d_in[4];
  const float* kU_b = (const float*)d_in[5];
  const float* vU_w = (const float*)d_in[6];
  const float* vU_b = (const float*)d_in[7];
  const float* qU_w = (const float*)d_in[8];
  const float* qU_b = (const float*)d_in[9];
  const float* kL_w = (const float*)d_in[10];
  const float* kL_b = (const float*)d_in[11];
  const float* vL_w = (const float*)d_in[12];
  const float* vL_b = (const float*)d_in[13];
  const float* beta = (const float*)d_in[14];
  float* out = (float*)d_out;

  char* ws = (char*)d_ws;
  // qbuf 512KB | kbuf 512KB | invZ 64KB | vbuf 8MB bf16 | E 128MB bf16
  float*  qbuf = (float*)(ws);
  float*  kbuf = (float*)(ws + (512u << 10));
  float*  invZ = (float*)(ws + (1024u << 10));
  __bf16* vbuf = (__bf16*)(ws + (1088u << 10));
  __bf16* Ebuf = (__bf16*)(ws + (1088u << 10) + (8u << 20));

  const size_t CN = (size_t)Ccfg * Ncfg;  // per-batch feature size
  const size_t NR = (size_t)Ncfg * Rcfg;
  dim3 blk(256);

  // Instances: 0:(LU,b0) 1:(LU,b1) 2:(UL,b0) 3:(UL,b1)
  // LU: q<-fL (qL), k<-fU (kU), v<-fU (vU) ; UL: q<-fU (qU), k<-fL (kL), v<-fL (vL)
  struct Job { const float *X, *W, *Bi; float* O; };
  const Job jobs[8] = {
      {fL,      qL_w, qL_b, qbuf + 0 * NR},
      {fL + CN, qL_w, qL_b, qbuf + 1 * NR},
      {fU,      qU_w, qU_b, qbuf + 2 * NR},
      {fU + CN, qU_w, qU_b, qbuf + 3 * NR},
      {fU,      kU_w, kU_b, kbuf + 0 * NR},
      {fU + CN, kU_w, kU_b, kbuf + 1 * NR},
      {fL,      kL_w, kL_b, kbuf + 2 * NR},
      {fL + CN, kL_w, kL_b, kbuf + 3 * NR},
  };
  for (int i = 0; i < 8; ++i)
    proj8_kernel<<<dim3(Ncfg / 256), blk, 0, stream>>>(jobs[i].X, jobs[i].W,
                                                       jobs[i].Bi, jobs[i].O);

  // E + Z for all instances (f32 WMMA on the logit GEMM).
  ekernel<<<dim3(Ncfg / 16, 4), blk, 0, stream>>>(qbuf, kbuf, Ebuf, invZ);

  // Grouped conv with 1/Z folded in.
  const float* vx[4]  = {fU, fU + CN, fL, fL + CN};
  const float* vw[4]  = {vU_w, vU_w, vL_w, vL_w};
  const float* vbi[4] = {vU_b, vU_b, vL_b, vL_b};
  for (int i = 0; i < 4; ++i)
    gconvz_kernel<<<dim3(Ncfg / 256, Ccfg), blk, 0, stream>>>(
        vx[i], vw[i], vbi[i], invZ + (size_t)i * Ncfg, vbuf + (size_t)i * CN);

  // Dominant GEMM: out = f + beta * (V x E^T).
  attn_gemm_kernel<<<dim3(Ncfg / 64, 4), blk, 0, stream>>>(vbuf, Ebuf, fL, fU,
                                                           out, beta);
}
